// WaveNet_49297634624165
// MI455X (gfx1250) — matrix operand order repaired and checked by reading, not compiled, not run on hardware
//
#include <hip/hip_runtime.h>
#include <hip/hip_bf16.h>

// ---------------------------------------------------------------------------
// WaveNet on gfx1250: bf16 WMMA (v_wmma_f32_16x16x32_bf16) for every GEMM.
// Activations stored time-major bf16 [t][32] so a B-fragment column is two
// contiguous b128 loads per lane. 1024-step zero guard handles dilation taps.
// Gated nonlinearity uses native CDNA5 V_TANH_F32 via inline asm.
// ---------------------------------------------------------------------------

typedef __attribute__((ext_vector_type(16))) __bf16 v16bf;
typedef __attribute__((ext_vector_type(8)))  __bf16 v8bf;
typedef __attribute__((ext_vector_type(8)))  float  v8f;

#define T_LEN  131072
#define RCH    32
#define SCH    16
#define QCH    256
#define NLAY   20
#define GUARD  1024   // >= max dilation (512)

// ---- native transcendental tanh (gfx1250 V_TANH_F32, TRANS pipe) ----------
// ISA: TRANS result needs 1 independent op / V_NOP before use; the compiler
// cannot see through inline asm, so the hazard slot is baked in.
__device__ __forceinline__ float tanh_hw(float x) {
    float r;
    asm volatile("v_tanh_f32 %0, %1\n\tv_nop" : "=v"(r) : "v"(x));
    return r;
}

// ---- fragment loaders (per documented 16-bit WMMA VGPR layouts) -----------

// B matrix (K x N = 32 x 16): lane n<16 holds column n, K={0..7,16..23};
// lane n+16 holds column n, K={8..15,24..31}. p = column base + 16*half.
__device__ __forceinline__ v16bf load_b16(const __bf16* p) {
    const int bh = (threadIdx.x >> 4) & 1;
    v8bf lo = *(const v8bf*)(p - 8 * bh);
    v8bf hi = *(const v8bf*)(p + 16 - 8 * bh);
    v16bf r;
#pragma unroll
    for (int i = 0; i < 8; ++i) { r[i] = lo[i]; r[8 + i] = hi[i]; }
    return r;
}

// A matrix (M x K = 16 x 32): lane L<16 holds row L, K={0..7,16..23};
// lane L+16 holds row L, K={8..15,24..31}. wrow points at row base (K contig).
__device__ __forceinline__ v16bf load_a16(const __bf16* wrow, int lh) {
    v8bf c0 = *(const v8bf*)(wrow + lh * 8);
    v8bf c1 = *(const v8bf*)(wrow + 16 + lh * 8);
    v16bf r;
#pragma unroll
    for (int i = 0; i < 8; ++i) { r[i] = c0[i]; r[8 + i] = c1[i]; }
    return r;
}

__device__ __forceinline__ v8f wmma_bf16(v16bf a, v16bf b, v8f c) {
    return __builtin_amdgcn_wmma_f32_16x16x32_bf16(
        /*neg_a=*/false, a, /*neg_b=*/false, b,
        /*c_mod=*/(short)0, c, /*reuse_a=*/false, /*reuse_b=*/false);
}

// ---- prep: repack fp32 weights -> bf16 GEMM layouts, zero skip & guards ---

__global__ void wn_prep(const float* __restrict__ wfilt, const float* __restrict__ wgate,
                        const float* __restrict__ wres,  const float* __restrict__ wskip,
                        const float* __restrict__ wfin1, const float* __restrict__ wfin2,
                        __bf16* __restrict__ xAg, __bf16* __restrict__ xBg,
                        float*  __restrict__ skip,
                        __bf16* __restrict__ wf, __bf16* __restrict__ wg,
                        __bf16* __restrict__ wr, __bf16* __restrict__ wsk,
                        __bf16* __restrict__ w1, __bf16* __restrict__ w2) {
    long i0 = blockIdx.x * (long)blockDim.x + threadIdx.x;
    long st = (long)gridDim.x * blockDim.x;
    for (long i = i0; i < (long)GUARD * RCH; i += st) {     // zero dilation guards
        xAg[i] = (__bf16)0.f; xBg[i] = (__bf16)0.f;
    }
    for (long i = i0; i < (long)T_LEN * SCH; i += st) skip[i] = 0.f;
    for (long i = i0; i < (long)NLAY * 2 * 32 * 32; i += st) {
        long c = i & 31, o = (i >> 5) & 31, tap = (i >> 10) & 1, l = i >> 11;
        // source layout (layer, D, R, K): tap interleaved last
        float fv = wfilt[((l * 32 + o) * 32 + c) * 2 + tap];
        float gv = wgate[((l * 32 + o) * 32 + c) * 2 + tap];
        wf[i] = (__bf16)fv; wg[i] = (__bf16)gv;
    }
    for (long i = i0; i < (long)NLAY * 32 * 32; i += st) wr[i]  = (__bf16)wres[i];
    for (long i = i0; i < (long)NLAY * 16 * 32; i += st) wsk[i] = (__bf16)wskip[i];
    for (long i = i0; i < (long)QCH * 32; i += st) {        // final1, K padded 16->32
        long c = i & 31, o = i >> 5;
        w1[i] = (c < 16) ? (__bf16)wfin1[o * 16 + c] : (__bf16)0.f;
    }
    for (long i = i0; i < (long)QCH * QCH; i += st) w2[i] = (__bf16)wfin2[i];
}

// ---- init 1x1 conv: (1,1,T) -> bf16 [t][32] ------------------------------

__global__ void wn_init(const float* __restrict__ x, const float* __restrict__ w_init,
                        __bf16* __restrict__ x0) {
    int t = blockIdx.x * blockDim.x + threadIdx.x;
    if (t >= T_LEN) return;
    float xv = x[t];
#pragma unroll
    for (int r = 0; r < RCH; ++r) x0[(long)t * RCH + r] = (__bf16)(xv * w_init[r]);
}

// ---- one gated dilated-conv layer: 11 WMMAs per 16-step tile per wave ----

__global__ __launch_bounds__(256) void wn_layer(
    const __bf16* __restrict__ xin,   // t=0 base; GUARD zeros behind it
    __bf16* __restrict__ xout,
    float*  __restrict__ skipacc,     // [t][16] fp32
    const __bf16* __restrict__ wfl,   // [2][32][32] (tap-major)
    const __bf16* __restrict__ wgl,
    const __bf16* __restrict__ wrl,   // [32][32]
    const __bf16* __restrict__ wskl,  // [16][32]
    int d) {
    __shared__ __attribute__((aligned(16))) __bf16 stage[8][16 * 40];

    const int lane = threadIdx.x & 31;
    const int wid  = threadIdx.x >> 5;
    const int t0   = (blockIdx.x * 8 + wid) * 16;
    const int lc   = lane & 15;        // column (time) / row (weight M)
    const int lh   = lane >> 4;        // K half select

    if (t0 + 256 < T_LEN)              // global_prefetch_b8 of next tiles
        __builtin_prefetch(xin + (long)(t0 + 256) * RCH, 0, 0);

    // activation B fragments: dilated tap (t-d) and current tap (t)
    v16bf Bm = load_b16(xin + (long)(t0 - d + lc) * RCH + lh * 16);
    v16bf B0 = load_b16(xin + (long)(t0 + lc) * RCH + lh * 16);

    v8f zero = {};
    v8f hF[2], hG[2];
#pragma unroll
    for (int mt = 0; mt < 2; ++mt) {   // out-channel tiles 0..15 / 16..31
        const __bf16* fr = wfl + (long)(mt * 16 + lc) * 32;
        const __bf16* gr = wgl + (long)(mt * 16 + lc) * 32;
        v8f af = wmma_bf16(load_a16(fr, lh),              Bm, zero);
        af     = wmma_bf16(load_a16(fr + 32 * 32, lh),    B0, af);
        v8f ag = wmma_bf16(load_a16(gr, lh),              Bm, zero);
        ag     = wmma_bf16(load_a16(gr + 32 * 32, lh),    B0, ag);
        hF[mt] = af; hG[mt] = ag;
    }

    // gated activation (native v_tanh_f32), restage via LDS into B layout
#pragma unroll
    for (int mt = 0; mt < 2; ++mt) {
        v8bf hb;
#pragma unroll
        for (int r = 0; r < 8; ++r)
            hb[r] = (__bf16)(tanh_hw(hF[mt][r]) * tanh_hw(hG[mt][r]));
        // C layout: lane holds channels mt*16 + lh*8 + r at time t0+lc
        *(v8bf*)&stage[wid][lc * 40 + mt * 16 + lh * 8] = hb;
    }
    __syncthreads();

    v16bf Bh = load_b16(&stage[wid][lc * 40 + lh * 16]);

    // skip 1x1 (M=16) with fp32 accumulate into global skip buffer
    {
        v8f as = wmma_bf16(load_a16(wskl + (long)lc * 32, lh), Bh, zero);
        float* sp = skipacc + (long)(t0 + lc) * SCH + lh * 8;
#pragma unroll
        for (int r = 0; r < 8; ++r) sp[r] += as[r];
    }

    // residual 1x1 (M=32) + skip-connection add, store next activations
#pragma unroll
    for (int mt = 0; mt < 2; ++mt) {
        v8f ar = wmma_bf16(load_a16(wrl + (long)(mt * 16 + lc) * 32, lh), Bh, zero);
        const long off = (long)(t0 + lc) * RCH + mt * 16 + lh * 8;
        v8bf xo = *(const v8bf*)(xin + off);
        v8bf nb;
#pragma unroll
        for (int r = 0; r < 8; ++r) nb[r] = (__bf16)(ar[r] + (float)xo[r]);
        *(v8bf*)(xout + off) = nb;
    }
}

// ---- fused head: relu(skip) -> 256x16 -> relu -> 256x256 -----------------

__global__ __launch_bounds__(128) void wn_final(
    const float* __restrict__ skip, const __bf16* __restrict__ w1,
    const __bf16* __restrict__ w2, float* __restrict__ out,
    int t_out, int t_start) {
    __shared__ __attribute__((aligned(16))) __bf16 hstage[4][16 * 40];
    __shared__ __attribute__((aligned(16))) __bf16 h1stage[4][16 * 264];

    const int lane = threadIdx.x & 31;
    const int wid  = threadIdx.x >> 5;
    const int t0   = (blockIdx.x * 4 + wid) * 16;
    const int lc   = lane & 15;
    const int lh   = lane >> 4;
    const int tloc = t0 + lc;
    const int trd  = t_start + (tloc < t_out ? tloc : t_out - 1);  // clamped read

    // stage relu(skip) as bf16, K padded 16 -> 32 with zeros
    {
        const float* sp = skip + (long)trd * SCH + lh * 8;
        v8bf hb, zz = {};
#pragma unroll
        for (int r = 0; r < 8; ++r) { float v = sp[r]; hb[r] = (__bf16)(v > 0.f ? v : 0.f); }
        *(v8bf*)&hstage[wid][lc * 40 + lh * 8]      = hb;
        *(v8bf*)&hstage[wid][lc * 40 + 16 + lh * 8] = zz;
    }
    __syncthreads();

    v16bf Bh = load_b16(&hstage[wid][lc * 40 + lh * 16]);
    v8f zero = {};

    // h1 = relu(W1 @ h0): 16 M-tiles of 256 outputs
    for (int mt = 0; mt < 16; ++mt) {
        v8f acc = wmma_bf16(load_a16(w1 + (long)(mt * 16 + lc) * 32, lh), Bh, zero);
        v8bf ob;
#pragma unroll
        for (int r = 0; r < 8; ++r) { float v = acc[r]; ob[r] = (__bf16)(v > 0.f ? v : 0.f); }
        *(v8bf*)&h1stage[wid][lc * 264 + mt * 16 + lh * 8] = ob;
    }
    __syncthreads();

    // out = W2 @ h1: 16 M-tiles x 8 K-steps of v_wmma_f32_16x16x32_bf16
    for (int mt = 0; mt < 16; ++mt) {
        v8f acc = {};
#pragma unroll
        for (int ks = 0; ks < 8; ++ks) {
            v16bf A = load_a16(w2 + (long)(mt * 16 + lc) * QCH + ks * 32, lh);
            v16bf B = load_b16(&h1stage[wid][lc * 264 + ks * 32 + lh * 16]);
            acc = wmma_bf16(A, B, acc);
        }
        if (tloc < t_out) {
#pragma unroll
            for (int r = 0; r < 8; ++r)
                out[(long)(mt * 16 + lh * 8 + r) * t_out + tloc] = acc[r];
        }
    }
}

// ---------------------------------------------------------------------------

extern "C" void kernel_launch(void* const* d_in, const int* in_sizes, int n_in,
                              void* d_out, int out_size, void* d_ws, size_t ws_size,
                              hipStream_t stream) {
    const float* x      = (const float*)d_in[0];
    const float* w_init = (const float*)d_in[1];
    const float* w_filt = (const float*)d_in[2];
    const float* w_gate = (const float*)d_in[3];
    const float* w_res  = (const float*)d_in[4];
    const float* w_skip = (const float*)d_in[5];
    const float* w_f1   = (const float*)d_in[6];
    const float* w_f2   = (const float*)d_in[7];
    float* out = (float*)d_out;

    // carve workspace
    char* p = (char*)d_ws;
    auto carve = [&](size_t bytes) { char* r = p; p += (bytes + 255) & ~(size_t)255; return r; };
    __bf16* xAg  = (__bf16*)carve((size_t)(GUARD + T_LEN) * RCH * 2);
    __bf16* xBg  = (__bf16*)carve((size_t)(GUARD + T_LEN) * RCH * 2);
    float*  skip = (float*) carve((size_t)T_LEN * SCH * 4);
    __bf16* wf   = (__bf16*)carve((size_t)NLAY * 2 * 32 * 32 * 2);
    __bf16* wg   = (__bf16*)carve((size_t)NLAY * 2 * 32 * 32 * 2);
    __bf16* wr   = (__bf16*)carve((size_t)NLAY * 32 * 32 * 2);
    __bf16* wsk  = (__bf16*)carve((size_t)NLAY * 16 * 32 * 2);
    __bf16* w1   = (__bf16*)carve((size_t)QCH * 32 * 2);
    __bf16* w2   = (__bf16*)carve((size_t)QCH * QCH * 2);
    (void)ws_size; (void)in_sizes; (void)n_in;

    __bf16* xA = xAg + (size_t)GUARD * RCH;   // t=0 bases (guard zeros behind)
    __bf16* xB = xBg + (size_t)GUARD * RCH;

    wn_prep<<<512, 256, 0, stream>>>(w_filt, w_gate, w_res, w_skip, w_f1, w_f2,
                                     xAg, xBg, skip, wf, wg, wr, wsk, w1, w2);
    wn_init<<<(T_LEN + 255) / 256, 256, 0, stream>>>(x, w_init, xA);

    __bf16* cur = xA;
    __bf16* nxt = xB;
    for (int idx = 0; idx < NLAY; ++idx) {
        int d = 1 << (idx % 10);
        wn_layer<<<T_LEN / 128, 256, 0, stream>>>(
            cur, nxt, skip,
            wf + (size_t)idx * 2 * 32 * 32, wg + (size_t)idx * 2 * 32 * 32,
            wr + (size_t)idx * 32 * 32,     wsk + (size_t)idx * 16 * 32, d);
        __bf16* t = cur; cur = nxt; nxt = t;
    }

    int t_out = out_size / QCH;
    if (t_out < 1) t_out = 1;
    int t_start = T_LEN - t_out; if (t_start < 0) t_start = 0;
    int ntiles = (t_out + 15) / 16;
    wn_final<<<(ntiles + 3) / 4, 128, 0, stream>>>(skip, w1, w2, out, t_out, t_start);
}
